// Attention_44049184588314
// MI455X (gfx1250) — matrix operand order repaired and checked by reading, not compiled, not run on hardware
//
#include <hip/hip_runtime.h>

// ---------------------------------------------------------------- constants
constexpr int B_      = 2;
constexpr int S_      = 2048;
constexpr int H_      = 2048;
constexpr int NH_     = 16;    // query heads
constexpr int NKV_    = 4;     // kv heads
constexpr int HD_     = 128;   // head dim
constexpr int THEADS_ = NH_ + 2 * NKV_;        // 24
constexpr int QKVO_   = THEADS_ * HD_;         // 3072
constexpr float QSCALE_ = 0.08838834764831845f; // 1/sqrt(128)

typedef __bf16 bf16;
typedef __attribute__((ext_vector_type(16))) __bf16 bf16x16;
typedef __attribute__((ext_vector_type(8)))  __bf16 bf16x8;
typedef __attribute__((ext_vector_type(8)))  float  f32x8;
typedef int v4i_ __attribute__((vector_size(16)));  // matches builtin param type

// ---------------------------------------------------------------- async copy
// CDNA5 async global->LDS (tracked by ASYNCcnt, no VGPR staging). Guarded so
// both host pass and older toolchains fall back to a synchronous copy.
#if __has_builtin(__builtin_amdgcn_global_load_async_to_lds_b128) && \
    __has_builtin(__builtin_amdgcn_s_wait_asynccnt)
#define AS1_ __attribute__((address_space(1)))
#define AS3_ __attribute__((address_space(3)))
#define ASYNC_CP_B128(dst, src)                                   \
  __builtin_amdgcn_global_load_async_to_lds_b128(                 \
      (AS1_ v4i_*)(AS1_ void*)(src),                              \
      (AS3_ v4i_*)(AS3_ void*)(dst), 0, 0)
#define ASYNC_WAIT(n) __builtin_amdgcn_s_wait_asynccnt(n)
#else
#define ASYNC_CP_B128(dst, src) (*(uint4*)(dst) = *(const uint4*)(src))
#define ASYNC_WAIT(n) ((void)0)
#endif

// ---------------------------------------------------------------- WMMA core
__device__ __forceinline__ f32x8 wmma_bf16(bf16x16 a, bf16x16 b, f32x8 c) {
  // D = A(16x32 bf16) * B(32x16 bf16) + C(16x16 f32)
  return __builtin_amdgcn_wmma_f32_16x16x32_bf16(
      /*neg_a=*/false, a, /*neg_b=*/false, b,
      /*c_mod=*/(short)0, c, /*reuse_a=*/false, /*reuse_b=*/false);
}

// A-matrix fragment (16x32 bf16). Contiguous k order used here (k = 16*hi + s):
//   lanes 0-15 : row M = lane,    elems = K[0..15]
//   lanes16-31 : row M = lane-16, elems = K[16..31]
// Caller passes p = &row_base[hi*16]  (hi = lane>>4).
__device__ __forceinline__ bf16x16 load_fragA(const bf16* p) {
  bf16x8 lo = *(const bf16x8*)p;
  bf16x8 hh = *(const bf16x8*)(p + 8);
  return __builtin_shufflevector(lo, hh, 0, 1, 2, 3, 4, 5, 6, 7,
                                 8, 9, 10, 11, 12, 13, 14, 15);
}

// B-matrix fragment (32x16 bf16). Lanes 0-15 hold K=0..15 of col N=lane,
// lanes 16-31 hold K=16..31. Caller passes p = &col_base[hi*16].
__device__ __forceinline__ bf16x16 load_fragB(const bf16* p) {
  bf16x8 lo = *(const bf16x8*)p;
  bf16x8 hh = *(const bf16x8*)(p + 8);
  return __builtin_shufflevector(lo, hh, 0, 1, 2, 3, 4, 5, 6, 7,
                                 8, 9, 10, 11, 12, 13, 14, 15);
}

// ---------------------------------------------------------------- convert
__global__ void f32_to_bf16_x4(const float* __restrict__ in,
                               bf16* __restrict__ out, size_t n4) {
  size_t i = blockIdx.x * (size_t)blockDim.x + threadIdx.x;
  if (i < n4) {
    float4 v = ((const float4*)in)[i];
    out[i * 4 + 0] = (bf16)v.x;
    out[i * 4 + 1] = (bf16)v.y;
    out[i * 4 + 2] = (bf16)v.z;
    out[i * 4 + 3] = (bf16)v.w;
  }
}

// ---------------------------------------------------------------- GEMM
// C[M,N] (f32) = A[M,K] (bf16, row-major) * B[N,K]^T (bf16, row-major)
// Block tile 128x256, 8 waves of 64x64 each (16 WMMAs per wave per K-step,
// 0.5 LDS fragment loads per WMMA). K step 32. Double-buffered LDS filled by
// async global->LDS copies (ASYNCcnt pipelined), last K-step peeled so the
// steady-state loop is branch-free.
__global__ __launch_bounds__(256) void gemm_bf16_wmma(
    const bf16* __restrict__ A, const bf16* __restrict__ Bm,
    float* __restrict__ C, int M, int N, int K) {
  __shared__ __attribute__((aligned(16))) bf16 As[2][128 * 32];  // 16 KB
  __shared__ __attribute__((aligned(16))) bf16 Bs[2][256 * 32];  // 32 KB

  const int tid  = threadIdx.x;
  const int wid  = tid >> 5;
  const int lane = tid & 31;
  const int hi   = lane >> 4;
  const int lm   = lane & 15;
  const int m_blk = blockIdx.y * 128;
  const int n_blk = blockIdx.x * 256;
  const int wm = (wid >> 2) * 64;  // wave row offset inside block tile
  const int wn = (wid & 3) * 64;   // wave col offset inside block tile

  // Stage a 128x32 A tile (512 b128 vecs) + 256x32 B tile (1024 b128 vecs):
  // each thread issues 2+4 async ops -> 6 ASYNCcnt tokens per wave per tile.
  auto stage = [&](int k0, int buf) {
#pragma unroll
    for (int v = tid; v < 512; v += 256) {
      const int r = v >> 2;
      const int c = (v & 3) << 3;
      ASYNC_CP_B128(&As[buf][r * 32 + c],
                    &A[(size_t)(m_blk + r) * K + k0 + c]);
    }
#pragma unroll
    for (int v = tid; v < 1024; v += 256) {
      const int r = v >> 2;
      const int c = (v & 3) << 3;
      ASYNC_CP_B128(&Bs[buf][r * 32 + c],
                    &Bm[(size_t)(n_blk + r) * K + k0 + c]);
    }
  };

  const f32x8 zz = {0.f, 0.f, 0.f, 0.f, 0.f, 0.f, 0.f, 0.f};
  f32x8 acc[4][4];
#pragma unroll
  for (int i = 0; i < 4; ++i)
#pragma unroll
    for (int j = 0; j < 4; ++j) acc[i][j] = zz;

  auto compute = [&](int cur) {
    bf16x16 af[4], bfr[4];
#pragma unroll
    for (int i = 0; i < 4; ++i)
      af[i] = load_fragA(&As[cur][(wm + i * 16 + lm) * 32 + hi * 16]);
#pragma unroll
    for (int j = 0; j < 4; ++j)
      bfr[j] = load_fragB(&Bs[cur][(wn + j * 16 + lm) * 32 + hi * 16]);
#pragma unroll
    for (int i = 0; i < 4; ++i)
#pragma unroll
      for (int j = 0; j < 4; ++j)
        acc[i][j] = wmma_bf16(af[i], bfr[j], acc[i][j]);
  };

  const int T = K / 32;
  stage(0, 0);
  int t = 0;
  for (; t < T - 1; ++t) {  // steady state: branch-free wait(6)
    stage((t + 1) * 32, (t & 1) ^ 1);  // overlap next tile's copy w/ compute
    ASYNC_WAIT(6);  // in-order async: tokens beyond 6 belong to next tile
    __syncthreads();
    compute(t & 1);
    __syncthreads();  // done reading; buffer may be overwritten next iter
  }
  ASYNC_WAIT(0);  // peeled last K-step
  __syncthreads();
  compute(t & 1);

  // D layout: reg r -> row (r + 8*hi), col lm within each 16x16 tile.
#pragma unroll
  for (int i = 0; i < 4; ++i)
#pragma unroll
    for (int j = 0; j < 4; ++j)
#pragma unroll
      for (int r = 0; r < 8; ++r) {
        const int row = m_blk + wm + i * 16 + r + 8 * hi;
        const int col = n_blk + wn + j * 16 + lm;
        C[(size_t)row * N + col] = acc[i][j][r];
      }
}

// ---------------------------------------------------------------- RoPE + scatter
// qkv f32 [B,S,24,128] -> q bf16 [B,16,S,128] (pre-scaled by 1/sqrt(HD)),
//                         k bf16 [B,4,S,128], vT bf16 [B,4,128,S]
__global__ void rope_scatter(const float* __restrict__ qkv,
                             const float* __restrict__ cosb,
                             const float* __restrict__ sinb,
                             bf16* __restrict__ q, bf16* __restrict__ k,
                             bf16* __restrict__ vt) {
  const size_t total = (size_t)B_ * S_ * THEADS_ * HD_;
  for (size_t x = blockIdx.x * (size_t)blockDim.x + threadIdx.x; x < total;
       x += (size_t)gridDim.x * blockDim.x) {
    const int d = (int)(x & (HD_ - 1));
    size_t t = x >> 7;
    const int head = (int)(t % THEADS_);
    t /= THEADS_;
    const int s = (int)(t % S_);
    const int bb = (int)(t / S_);
    float val = qkv[x];
    if (head < NH_ + NKV_) {  // RoPE for q and k heads
      const float partner = qkv[x + (d < 64 ? 64 : -64)];
      const float rh = (d < 64) ? -partner : partner;
      val = val * cosb[s * HD_ + d] + rh * sinb[s * HD_ + d];
    }
    if (head < NH_) {
      q[(((size_t)bb * NH_ + head) * S_ + s) * HD_ + d] = (bf16)(val * QSCALE_);
    } else if (head < NH_ + NKV_) {
      k[(((size_t)bb * NKV_ + (head - NH_)) * S_ + s) * HD_ + d] = (bf16)val;
    } else {
      vt[(((size_t)bb * NKV_ + (head - NH_ - NKV_)) * HD_ + d) * S_ + s] =
          (bf16)val;
    }
  }
}

// ---------------------------------------------------------------- flash attention
// grid (S/128, NH, B), 8 waves/block; each wave owns 16 query rows.
__global__ __launch_bounds__(256) void flash_attn_wmma(
    const bf16* __restrict__ q, const bf16* __restrict__ k,
    const bf16* __restrict__ vt, bf16* __restrict__ attn_out) {
  __shared__ __attribute__((aligned(16))) bf16 Plds[8][16 * 32];

  const int tid = threadIdx.x, wid = tid >> 5, lane = tid & 31;
  const int hi = lane >> 4, lm = lane & 15;
  const int h = blockIdx.y, b = blockIdx.z;
  const int m0 = blockIdx.x * 128 + wid * 16;  // first query row of this wave
  const int kvh = h >> 2;                      // 4 q-heads per kv-head

  const bf16* qbase = q + (size_t)(b * NH_ + h) * S_ * HD_;
  const bf16* kbase = k + (size_t)(b * NKV_ + kvh) * S_ * HD_;
  const bf16* vbase = vt + (size_t)(b * NKV_ + kvh) * HD_ * S_;

  // Q fragments (A operand), held for the whole key loop.
  bf16x16 qf[4];
#pragma unroll
  for (int c = 0; c < 4; ++c)
    qf[c] = load_fragA(qbase + (size_t)(m0 + lm) * HD_ + c * 32 + hi * 16);

  const f32x8 zz = {0.f, 0.f, 0.f, 0.f, 0.f, 0.f, 0.f, 0.f};
  f32x8 o[8];
#pragma unroll
  for (int j = 0; j < 8; ++j) o[j] = zz;
  float mrow[8], lrow[8];
#pragma unroll
  for (int r = 0; r < 8; ++r) { mrow[r] = -1e30f; lrow[r] = 0.f; }

  bf16* pl = &Plds[wid][0];  // wave-private 16x32 P tile
  const int kend = m0 + 16;  // causal bound (exclusive)

  for (int k0 = 0; k0 < kend; k0 += 32) {
    // ---- scores: two 16x16 tiles over 32 keys, 4 K-chunks of head dim
    f32x8 s0 = zz, s1 = zz;
#pragma unroll
    for (int c = 0; c < 4; ++c) {
      bf16x16 kf0 =
          load_fragB(kbase + (size_t)(k0 + lm) * HD_ + c * 32 + hi * 16);
      bf16x16 kf1 =
          load_fragB(kbase + (size_t)(k0 + 16 + lm) * HD_ + c * 32 + hi * 16);
      s0 = wmma_bf16(qf[c], kf0, s0);
      s1 = wmma_bf16(qf[c], kf1, s1);
    }

    // ---- causal mask + online softmax (reg r -> row m0 + r + 8*hi, col lm)
    const int row0 = m0 + 8 * hi;
    const int col0 = k0 + lm, col1 = col0 + 16;
#pragma unroll
    for (int r = 0; r < 8; ++r) {
      const int rg = row0 + r;
      float v0 = (col0 <= rg) ? s0[r] : -1e30f;
      float v1 = (col1 <= rg) ? s1[r] : -1e30f;
      float mx = fmaxf(v0, v1);
#pragma unroll
      for (int off = 1; off < 16; off <<= 1)
        mx = fmaxf(mx, __shfl_xor(mx, off));
      const float mnew = fmaxf(mrow[r], mx);
      const float corr = __expf(mrow[r] - mnew);
      mrow[r] = mnew;
      const float p0 = __expf(v0 - mnew);
      const float p1 = __expf(v1 - mnew);
      float rs = p0 + p1;
#pragma unroll
      for (int off = 1; off < 16; off <<= 1) rs += __shfl_xor(rs, off);
      lrow[r] = lrow[r] * corr + rs;
#pragma unroll
      for (int j = 0; j < 8; ++j) o[j][r] *= corr;
      // stash P as bf16 in LDS for re-fragmenting as an A operand
      pl[(r + 8 * hi) * 32 + lm] = (bf16)p0;
      pl[(r + 8 * hi) * 32 + 16 + lm] = (bf16)p1;
    }

    // ---- O += P(16x32) * V(32x128): 8 WMMAs across head dim
    bf16x16 pf = load_fragA(pl + lm * 32 + hi * 16);
#pragma unroll
    for (int j = 0; j < 8; ++j) {
      bf16x16 vf =
          load_fragB(vbase + (size_t)(j * 16 + lm) * S_ + k0 + hi * 16);
      o[j] = wmma_bf16(pf, vf, o[j]);
    }
  }

  // ---- epilogue: normalize and store bf16 [B,S,NH*HD]
#pragma unroll
  for (int r = 0; r < 8; ++r) {
    const float invl = 1.0f / lrow[r];
    const int srow = m0 + r + 8 * hi;
    bf16* orow = attn_out + ((size_t)(b * S_ + srow)) * (NH_ * HD_) + h * HD_;
#pragma unroll
    for (int j = 0; j < 8; ++j) orow[j * 16 + lm] = (bf16)(o[j][r] * invl);
  }
}

// ---------------------------------------------------------------- launch
extern "C" void kernel_launch(void* const* d_in, const int* in_sizes, int n_in,
                              void* d_out, int out_size, void* d_ws,
                              size_t ws_size, hipStream_t stream) {
  const float* hs   = (const float*)d_in[0];
  const float* cosb = (const float*)d_in[1];
  const float* sinb = (const float*)d_in[2];
  const float* wqkv = (const float*)d_in[3];
  const float* wo   = (const float*)d_in[4];
  float* out = (float*)d_out;

  char* p = (char*)d_ws;
  auto alloc = [&](size_t bytes) {
    char* r = p;
    p += (bytes + 255) & ~(size_t)255;
    return r;
  };
  const size_t n_hs   = (size_t)B_ * S_ * H_;      // 8.4M
  const size_t n_wqkv = (size_t)QKVO_ * H_;        // 6.3M
  const size_t n_wo   = (size_t)H_ * (NH_ * HD_);  // 4.2M

  bf16*  hs_b   = (bf16*)alloc(n_hs * 2);
  bf16*  wqkv_b = (bf16*)alloc(n_wqkv * 2);
  bf16*  wo_b   = (bf16*)alloc(n_wo * 2);
  float* qkv    = (float*)alloc((size_t)B_ * S_ * QKVO_ * 4);
  bf16*  qb     = (bf16*)alloc((size_t)B_ * NH_ * S_ * HD_ * 2);
  bf16*  kb     = (bf16*)alloc((size_t)B_ * NKV_ * S_ * HD_ * 2);
  bf16*  vtb    = (bf16*)alloc((size_t)B_ * NKV_ * HD_ * S_ * 2);
  bf16*  attnb  = (bf16*)alloc((size_t)B_ * S_ * NH_ * HD_ * 2);

  // 1) downcast operands to bf16
  f32_to_bf16_x4<<<(unsigned)((n_hs / 4 + 255) / 256), 256, 0, stream>>>(
      hs, hs_b, n_hs / 4);
  f32_to_bf16_x4<<<(unsigned)((n_wqkv / 4 + 255) / 256), 256, 0, stream>>>(
      wqkv, wqkv_b, n_wqkv / 4);
  f32_to_bf16_x4<<<(unsigned)((n_wo / 4 + 255) / 256), 256, 0, stream>>>(
      wo, wo_b, n_wo / 4);

  // 2) QKV projection: [B*S, H] x [QKVO, H]^T -> qkv f32
  dim3 g1(QKVO_ / 256, (B_ * S_) / 128);
  gemm_bf16_wmma<<<g1, 256, 0, stream>>>(hs_b, wqkv_b, qkv, B_ * S_, QKVO_, H_);

  // 3) RoPE + scatter into q/k/vT bf16 layouts
  rope_scatter<<<8192, 256, 0, stream>>>(qkv, cosb, sinb, qb, kb, vtb);

  // 4) causal GQA flash attention
  dim3 g2(S_ / 128, NH_, B_);
  flash_attn_wmma<<<g2, 256, 0, stream>>>(qb, kb, vtb, attnb);

  // 5) output projection: [B*S, 2048] x [H, 2048]^T -> d_out f32
  dim3 g3((NH_ * HD_) / 256, (B_ * S_) / 128);
  gemm_bf16_wmma<<<g3, 256, 0, stream>>>(attnb, wo_b, out, B_ * S_, NH_ * HD_,
                                         H_);
}
